// PromptFormer_77653008711745
// MI455X (gfx1250) — matrix operand order repaired and checked by reading, not compiled, not run on hardware
//
#include <hip/hip_runtime.h>
#include <hip/hip_bf16.h>

typedef __attribute__((ext_vector_type(16))) _Float16 v16h;
typedef __attribute__((ext_vector_type(8)))  float    v8f;

#define WMMA_F16(a, b, c) \
    __builtin_amdgcn_wmma_f32_16x16x32_f16(false, (a), false, (b), (short)0, (c), false, false)

// ---------------------------------------------------------------- helpers ---

__device__ __forceinline__ float gelu_exact(float v) {
    return 0.5f * v * (1.0f + erff(v * 0.70710678118654752440f));
}

// A fragment 16(M)x32(K) gathered from row-major f32 (cold path: H only)
__device__ __forceinline__ v16h load_a_frag(const float* __restrict__ src, int lda, int lane) {
    const int m  = lane & 15;
    const int kb = (lane >> 4) << 3;
    v16h a;
#pragma unroll
    for (int j = 0; j < 16; ++j) {
        const int k = kb + (j & 7) + ((j >> 3) << 4);
        a[j] = (_Float16)src[(size_t)m * lda + k];
    }
    return a;
}

// ------------------------------------------------------------- row rnorm ---
__global__ __launch_bounds__(256) void k_rownorm(const float* __restrict__ src,
                                                 float* __restrict__ inv_out,
                                                 int nrows) {
    const int row = blockIdx.x * 8 + (threadIdx.x >> 5);
    if (row >= nrows) return;
    const int lane = threadIdx.x & 31;
    const float* r = src + (size_t)row * 128;
    float v0 = r[lane], v1 = r[lane + 32], v2 = r[lane + 64], v3 = r[lane + 96];
    float ss = v0 * v0 + v1 * v1 + v2 * v2 + v3 * v3;
    ss += __shfl_xor(ss, 1, 32);
    ss += __shfl_xor(ss, 2, 32);
    ss += __shfl_xor(ss, 4, 32);
    ss += __shfl_xor(ss, 8, 32);
    ss += __shfl_xor(ss, 16, 32);
    if (lane == 0) inv_out[row] = rsqrtf(ss + 1e-30f);
}

// ------------------------------------------------- operand packing kernels ---
// Pack f32 row-major [rows x 128] into f16 A-fragment blobs.
// fragid = mt*4 + kt ; blob element = dst[fragid*32 + lane] (v16h, 32B, coalesced)
__global__ __launch_bounds__(256) void k_pack_a(const float* __restrict__ src,
                                                v16h* __restrict__ dst) {
    const int lane = threadIdx.x & 31;
    const int wid = blockIdx.x * 8 + (threadIdx.x >> 5);
    const int mt = wid >> 2, kt = wid & 3;
    const int m = lane & 15, h = lane >> 4;
    v16h a;
#pragma unroll
    for (int j = 0; j < 16; ++j) {
        const int k = kt * 32 + (h << 3) + (j & 7) + ((j >> 3) << 4);
        a[j] = (_Float16)src[(size_t)(mt * 16 + m) * 128 + k];
    }
    dst[(size_t)wid * 32 + lane] = a;
}

// Pack x^T A-fragments for proto stage 1 (scatter once instead of 8x in-loop).
// fragid = (b*8 + mt)*32 + kc ; elem(M=d, K=s) = x[b][s][d]
__global__ __launch_bounds__(256) void k_pack_at(const float* __restrict__ x,
                                                 v16h* __restrict__ dst) {
    const int lane = threadIdx.x & 31;
    const int wid = blockIdx.x * 8 + (threadIdx.x >> 5);   // 0..4095
    const int kc = wid & 31;
    const int mt = (wid >> 5) & 7;
    const int b  = wid >> 8;
    const int m = lane & 15, h = lane >> 4;
    v16h a;
#pragma unroll
    for (int j = 0; j < 16; ++j) {
        const int s = kc * 32 + (h << 3) + (j & 7) + ((j >> 3) << 4);
        a[j] = (_Float16)x[(size_t)b * 131072 + (size_t)s * 128 + mt * 16 + m];
    }
    dst[(size_t)wid * 32 + lane] = a;
}

// Pack f32 row-major [K x 128] into f16 B-fragment blobs. fragid = kc*8 + nt.
__global__ __launch_bounds__(256) void k_pack_b(const float* __restrict__ src,
                                                v16h* __restrict__ dst) {
    const int lane = threadIdx.x & 31;
    const int wid = blockIdx.x * 8 + (threadIdx.x >> 5);
    const int kc = wid >> 3, nt = wid & 7;
    const int n = lane & 15, h = lane >> 4;
    v16h b;
#pragma unroll
    for (int j = 0; j < 16; ++j) {
        const int k = kc * 32 + (h << 3) + (j & 7) + ((j >> 3) << 4);
        b[j] = (_Float16)src[(size_t)k * 128 + nt * 16 + n];
    }
    dst[(size_t)wid * 32 + lane] = b;
}

// Pack normalized prototypes as B^T fragments for S = Q @ pn^T.
// fragid = (kc*2 + n2)*4 + kt ; elem(K=d, N=key) = proto[key][d] * pinv[key]
__global__ __launch_bounds__(256) void k_pack_pn(const float* __restrict__ proto,
                                                 const float* __restrict__ pinv,
                                                 v16h* __restrict__ dst) {
    const int lane = threadIdx.x & 31;
    const int wid = blockIdx.x * 8 + (threadIdx.x >> 5);
    const int kt = wid & 3, n2 = (wid >> 2) & 1, kc = wid >> 3;
    const int n = lane & 15, h = lane >> 4;
    const int key = kc * 32 + n2 * 16 + n;
    const float inv = pinv[key];
    v16h b;
#pragma unroll
    for (int j = 0; j < 16; ++j) {
        const int d = kt * 32 + (h << 3) + (j & 7) + ((j >> 3) << 4);
        b[j] = (_Float16)(proto[(size_t)key * 128 + d] * inv);
    }
    dst[(size_t)wid * 32 + lane] = b;
}

// ------------------------------------------- proto stage 1: H = gelu(xT W) ---
// All-wide-load inner loop: A from packed x^T frags, B from packed w_n1 frags.
__global__ __launch_bounds__(256) void k_proto_h(const v16h* __restrict__ xTA,
                                                 const v16h* __restrict__ wn1B,
                                                 const float* __restrict__ b_n1,
                                                 float* __restrict__ H) {
    const int lane = threadIdx.x & 31;
    const int tile = blockIdx.x * 8 + (threadIdx.x >> 5);   // 0..1023
    const int b  = tile >> 6;
    const int mt = (tile >> 3) & 7;
    const int nt = tile & 7;
    const int m = lane & 15, hgrp = lane >> 4;
    v8f acc = {};
    for (int kc = 0; kc < 32; ++kc) {
        v16h a  = xTA[(size_t)((b * 8 + mt) * 32 + kc) * 32 + lane];
        v16h bf = wn1B[(size_t)(kc * 8 + nt) * 32 + lane];
        acc = WMMA_F16(a, bf, acc);
    }
#pragma unroll
    for (int v = 0; v < 8; ++v) {
        const int row = mt * 16 + v + (hgrp << 3);
        const int col = nt * 16 + m;
        H[((size_t)b * 128 + row) * 128 + col] = gelu_exact(acc[v] + b_n1[col]);
    }
}

// ---------------------------- proto stage 2: proto = H W2 + b, store as BNxD ---
__global__ __launch_bounds__(256) void k_proto_out(const float* __restrict__ H,
                                                   const v16h* __restrict__ wn2B,
                                                   const float* __restrict__ b_n2,
                                                   float* __restrict__ proto_out) {
    const int lane = threadIdx.x & 31;
    const int tile = blockIdx.x * 8 + (threadIdx.x >> 5);   // 0..1023
    const int b  = tile >> 6;
    const int mt = (tile >> 3) & 7;
    const int nt = tile & 7;
    const int m = lane & 15, hgrp = lane >> 4;
    v8f acc = {};
#pragma unroll
    for (int kt = 0; kt < 4; ++kt) {
        v16h a  = load_a_frag(H + ((size_t)b * 128 + mt * 16) * 128 + kt * 32, 128, lane);
        v16h bf = wn2B[(size_t)(kt * 8 + nt) * 32 + lane];
        acc = WMMA_F16(a, bf, acc);
    }
#pragma unroll
    for (int v = 0; v < 8; ++v) {
        const int d  = mt * 16 + v + (hgrp << 3);
        const int nn = nt * 16 + m;
        proto_out[((size_t)b * 128 + nn) * 128 + d] = acc[v] + b_n2[nn];
    }
}

// --------------------------------------------------- flash attention core ---
// One wave handles M=32 query rows. Output written directly as packed f16
// A-fragments (LDS transpose) -- no f32 round trip for `prompt`.
__global__ __launch_bounds__(32) void k_flash(const v16h* __restrict__ xA,
                                              const float* __restrict__ xinv,
                                              const v16h* __restrict__ pnP,
                                              const v16h* __restrict__ vP,
                                              v16h* __restrict__ promptA) {
    __shared__ _Float16 lds_p[2048];   // P repack (2KB) + finalize 16x128 tile (4KB)
    const int lane = threadIdx.x & 31;
    const int m = lane & 15, hgrp = lane >> 4;
    const int mt0 = blockIdx.x * 2;

    v16h qa[2][4];
#pragma unroll
    for (int t = 0; t < 2; ++t) {
        const _Float16 hinv = (_Float16)xinv[(mt0 + t) * 16 + m];
#pragma unroll
        for (int kt = 0; kt < 4; ++kt) {
            v16h q = xA[(size_t)((mt0 + t) * 4 + kt) * 32 + lane];
            qa[t][kt] = q * hinv;
        }
    }

    v8f o[2][8];
    float mrun[2][8], lrun[2][8];
#pragma unroll
    for (int t = 0; t < 2; ++t)
#pragma unroll
        for (int v = 0; v < 8; ++v) {
            o[t][v] = (v8f){};
            mrun[t][v] = -1e30f;
            lrun[t][v] = 0.0f;
        }

    for (int kc = 0; kc < 64; ++kc) {
        if (kc + 1 < 64) {   // prefetch next chunk's packed fragments (L2 hot)
#pragma unroll
            for (int f = 0; f < 8; ++f) {
                __builtin_prefetch((const void*)(pnP + ((size_t)(kc + 1) * 8 + f) * 32 + lane), 0, 0);
                __builtin_prefetch((const void*)(vP  + ((size_t)(kc + 1) * 8 + f) * 32 + lane), 0, 0);
            }
        }
        // S = Q @ pn^T
        v16h pf[2][4];
#pragma unroll
        for (int n2 = 0; n2 < 2; ++n2)
#pragma unroll
            for (int kt = 0; kt < 4; ++kt)
                pf[n2][kt] = pnP[(size_t)((kc * 2 + n2) * 4 + kt) * 32 + lane];
        v8f s[2][2];
#pragma unroll
        for (int t = 0; t < 2; ++t)
#pragma unroll
            for (int n2 = 0; n2 < 2; ++n2) {
                v8f acc = {};
#pragma unroll
                for (int kt = 0; kt < 4; ++kt)
                    acc = WMMA_F16(qa[t][kt], pf[n2][kt], acc);
                s[t][n2] = acc;
            }
        // online softmax (row v+8*hgrp lives in one 16-lane half)
#pragma unroll
        for (int t = 0; t < 2; ++t)
#pragma unroll
            for (int v = 0; v < 8; ++v) {
                float tx = fmaxf(s[t][0][v], s[t][1][v]);
                tx = fmaxf(tx, __shfl_xor(tx, 1, 16));
                tx = fmaxf(tx, __shfl_xor(tx, 2, 16));
                tx = fmaxf(tx, __shfl_xor(tx, 4, 16));
                tx = fmaxf(tx, __shfl_xor(tx, 8, 16));
                const float mn = fmaxf(mrun[t][v], tx);
                const float corr = __expf(mrun[t][v] - mn);
                const float p0 = __expf(s[t][0][v] - mn);
                const float p1 = __expf(s[t][1][v] - mn);
                s[t][0][v] = p0; s[t][1][v] = p1;
                float rs = p0 + p1;
                rs += __shfl_xor(rs, 1, 16);
                rs += __shfl_xor(rs, 2, 16);
                rs += __shfl_xor(rs, 4, 16);
                rs += __shfl_xor(rs, 8, 16);
                lrun[t][v] = lrun[t][v] * corr + rs;
                mrun[t][v] = mn;
#pragma unroll
                for (int nf = 0; nf < 8; ++nf) o[t][nf][v] *= corr;
            }
        // repack P (C-layout) -> A-layout via LDS
#pragma unroll
        for (int t = 0; t < 2; ++t)
#pragma unroll
            for (int v = 0; v < 8; ++v) {
                const int row = v + (hgrp << 3);
                lds_p[t * 512 + row * 32 + m]      = (_Float16)s[t][0][v];
                lds_p[t * 512 + row * 32 + 16 + m] = (_Float16)s[t][1][v];
            }
        asm volatile("s_wait_dscnt 0" ::: "memory");
        v16h pa[2];
#pragma unroll
        for (int t = 0; t < 2; ++t)
#pragma unroll
            for (int j = 0; j < 16; ++j)
                pa[t][j] = lds_p[t * 512 + m * 32 + (hgrp << 3) + (j & 7) + ((j >> 3) << 4)];
        // O += P @ V (each V fragment reused by both row sub-tiles)
#pragma unroll
        for (int nf = 0; nf < 8; ++nf) {
            v16h vf = vP[(size_t)(kc * 8 + nf) * 32 + lane];
#pragma unroll
            for (int t = 0; t < 2; ++t)
                o[t][nf] = WMMA_F16(pa[t], vf, o[t][nf]);
        }
        asm volatile("" ::: "memory");
    }
    // finalize: divide by running sum; LDS-transpose each 16x128 tile into
    // packed A-fragments (consumed directly by the d-MLP).
#pragma unroll
    for (int t = 0; t < 2; ++t) {
        asm volatile("s_wait_dscnt 0" ::: "memory");   // guard LDS reuse
#pragma unroll
        for (int v = 0; v < 8; ++v) {
            const float inv = 1.0f / lrun[t][v];
            const int row = v + (hgrp << 3);
#pragma unroll
            for (int nf = 0; nf < 8; ++nf)
                lds_p[row * 128 + nf * 16 + m] = (_Float16)(o[t][nf][v] * inv);
        }
        asm volatile("s_wait_dscnt 0" ::: "memory");
#pragma unroll
        for (int kt = 0; kt < 4; ++kt) {
            v16h f;
#pragma unroll
            for (int j = 0; j < 16; ++j) {
                const int c = kt * 32 + (hgrp << 3) + (j & 7) + ((j >> 3) << 4);
                f[j] = lds_p[m * 128 + c];
            }
            promptA[(size_t)((mt0 + t) * 4 + kt) * 32 + lane] = f;
        }
    }
}

// ----- G = gelu(prompt @ w_d1 + b_d1), emitted directly as packed A-frags ----
// One wave computes a 16x32 output tile (2 accumulators) == one A-fragment's
// K-range, transposed through a per-wave LDS slice. No f32 round trip for G.
__global__ __launch_bounds__(256) void k_mlp_d1(const v16h* __restrict__ pA,
                                                const v16h* __restrict__ wB,
                                                const float* __restrict__ bias,
                                                v16h* __restrict__ GA) {
    __shared__ _Float16 lds[8][512];   // 16x32 f16 per wave
    const int lane = threadIdx.x & 31;
    const int warp = threadIdx.x >> 5;
    const int tile = blockIdx.x * 8 + warp;                 // 0..4095
    const int mt = tile >> 2, kt = tile & 3;
    const int m = lane & 15, hgrp = lane >> 4;
    v8f acc0 = {}, acc1 = {};
#pragma unroll
    for (int kk = 0; kk < 4; ++kk) {
        v16h a  = pA[(size_t)(mt * 4 + kk) * 32 + lane];
        v16h b0 = wB[(size_t)(kk * 8 + kt * 2) * 32 + lane];
        acc0 = WMMA_F16(a, b0, acc0);
        v16h b1 = wB[(size_t)(kk * 8 + kt * 2 + 1) * 32 + lane];
        acc1 = WMMA_F16(a, b1, acc1);
    }
#pragma unroll
    for (int v = 0; v < 8; ++v) {
        const int row = v + (hgrp << 3);
        const int c0 = kt * 32 + m, c1 = kt * 32 + 16 + m;
        lds[warp][row * 32 + m]      = (_Float16)gelu_exact(acc0[v] + bias[c0]);
        lds[warp][row * 32 + 16 + m] = (_Float16)gelu_exact(acc1[v] + bias[c1]);
    }
    asm volatile("s_wait_dscnt 0" ::: "memory");
    v16h f;
#pragma unroll
    for (int j = 0; j < 16; ++j)
        f[j] = lds[warp][m * 32 + (hgrp << 3) + (j & 7) + ((j >> 3) << 4)];
    GA[(size_t)(mt * 4 + kt) * 32 + lane] = f;
}

// -------- fused tail: res = x w_p + b_p ; main = G w_d2 + b_d2 + res ---------
__global__ __launch_bounds__(256) void k_final(const v16h* __restrict__ GA,
                                               const v16h* __restrict__ wd2B,
                                               const v16h* __restrict__ xA,
                                               const v16h* __restrict__ wpB,
                                               const float* __restrict__ b_d2,
                                               const float* __restrict__ b_p,
                                               float* __restrict__ out_main,
                                               float* __restrict__ out_res) {
    const int lane = threadIdx.x & 31;
    const int tile = blockIdx.x * 8 + (threadIdx.x >> 5);   // 0..8191
    const int mt = tile >> 3, nt = tile & 7;
    const int m = lane & 15, hgrp = lane >> 4;
    v8f acc1 = {}, acc2 = {};
#pragma unroll
    for (int kt = 0; kt < 4; ++kt) {
        v16h a1 = GA[(size_t)(mt * 4 + kt) * 32 + lane];
        v16h b1 = wd2B[(size_t)(kt * 8 + nt) * 32 + lane];
        acc1 = WMMA_F16(a1, b1, acc1);
        v16h a2 = xA[(size_t)(mt * 4 + kt) * 32 + lane];
        v16h b2 = wpB[(size_t)(kt * 8 + nt) * 32 + lane];
        acc2 = WMMA_F16(a2, b2, acc2);
    }
#pragma unroll
    for (int v = 0; v < 8; ++v) {
        const int row = mt * 16 + v + (hgrp << 3);
        const int col = nt * 16 + m;
        const float r = acc2[v] + b_p[col];
        out_res[(size_t)row * 128 + col]  = r;
        out_main[(size_t)row * 128 + col] = acc1[v] + b_d2[col] + r;
    }
}

// ----------------------------------------------------------------- launch ---
extern "C" void kernel_launch(void* const* d_in, const int* in_sizes, int n_in,
                              void* d_out, int out_size, void* d_ws, size_t ws_size,
                              hipStream_t stream) {
    (void)in_sizes; (void)n_in; (void)out_size; (void)ws_size;
    const float* x    = (const float*)d_in[0];
    // d_in[1] = umask: per-row-constant mask is a softmax no-op -> unused
    const float* w_n1 = (const float*)d_in[2];
    const float* b_n1 = (const float*)d_in[3];
    const float* w_n2 = (const float*)d_in[4];
    const float* b_n2 = (const float*)d_in[5];
    const float* w_d1 = (const float*)d_in[6];
    const float* b_d1 = (const float*)d_in[7];
    const float* w_d2 = (const float*)d_in[8];
    const float* b_d2 = (const float*)d_in[9];
    const float* w_p  = (const float*)d_in[10];
    const float* b_p  = (const float*)d_in[11];

    // ---- workspace: f32 region then 32B-aligned f16 fragment-blob region ----
    float* ws   = (float*)d_ws;
    float* xinv = ws;                   // 16384
    float* pinv = xinv + 16384;         // 2048
    float* H    = pinv + 2048;          // 262144
    // half region starts at float offset 280,576 (byte offset 1,122,304 = 32*35072)
    _Float16* hbase = (_Float16*)(H + 262144);
    v16h* xA      = (v16h*)hbase;                 // 4096 frags
    v16h* xTA     = (v16h*)(hbase + 2097152);     // 4096 frags (x^T for proto)
    v16h* promptA = (v16h*)(hbase + 2 * 2097152); // 4096 frags
    v16h* GA      = (v16h*)(hbase + 3 * 2097152); // 4096 frags
    v16h* pnP     = (v16h*)(hbase + 4 * 2097152); // 512 frags
    v16h* vP      = (v16h*)(hbase + 4 * 2097152 + 262144);      // 512 frags
    v16h* wn1B    = (v16h*)(hbase + 4 * 2097152 + 2 * 262144);  // 256 frags
    _Float16* wtail = hbase + 4 * 2097152 + 2 * 262144 + 131072;
    v16h* wn2B = (v16h*)wtail;                    // 32 frags each
    v16h* wd1B = (v16h*)(wtail + 16384);
    v16h* wd2B = (v16h*)(wtail + 2 * 16384);
    v16h* wpB  = (v16h*)(wtail + 3 * 16384);

    // output layout: [prompt+res (2097152)] [proto (262144)] [res (2097152)]
    float* out_main  = (float*)d_out;
    float* proto_out = out_main + 2097152;
    float* out_res   = proto_out + 262144;

    k_rownorm  <<<2048, 256, 0, stream>>>(x, xinv, 16384);
    k_pack_a   <<<512, 256, 0, stream>>>(x, xA);
    k_pack_at  <<<512, 256, 0, stream>>>(x, xTA);
    k_pack_b   <<<32, 256, 0, stream>>>(w_n1, wn1B);
    k_pack_b   <<<4, 256, 0, stream>>>(w_n2, wn2B);
    k_pack_b   <<<4, 256, 0, stream>>>(w_d1, wd1B);
    k_pack_b   <<<4, 256, 0, stream>>>(w_d2, wd2B);
    k_pack_b   <<<4, 256, 0, stream>>>(w_p, wpB);

    k_proto_h  <<<128, 256, 0, stream>>>(xTA, wn1B, b_n1, H);
    k_proto_out<<<128, 256, 0, stream>>>(H, wn2B, b_n2, proto_out);
    k_rownorm  <<<256, 256, 0, stream>>>(proto_out, pinv, 2048);
    k_pack_pn  <<<64, 256, 0, stream>>>(proto_out, pinv, pnP);
    k_pack_b   <<<64, 256, 0, stream>>>(proto_out, vP);   // V fragments

    k_flash    <<<512, 32, 0, stream>>>(xA, xinv, pnP, vP, promptA);
    k_mlp_d1   <<<512, 256, 0, stream>>>(promptA, wd1B, b_d1, GA);
    k_final    <<<1024, 256, 0, stream>>>(GA, wd2B, xA, wpB, b_d2, b_p, out_main, out_res);
}
